// NeuralMemory_32985348833383
// MI455X (gfx1250) — compile-verified, matrix operand order checked by reading, not run on hardware
//
#include <hip/hip_runtime.h>

// ---------------------------------------------------------------------------
// Neural-memory (test-time-training) forward for MI455X (gfx1250).
//   B=2, L=2048, D=M=2048, CHUNK=128 -> 16 sequential chunks.
// All GEMMs on v_wmma_f32_16x16x32_bf16 (fp32 accumulate). Producers write
// bf16 operand mirrors once; GEMMs stage bf16 tiles into LDS via CDNA5
// async global->LDS copies (global_load_async_to_lds_b128 + s_wait_asynccnt)
// for row-major operands, manual staging for transposed operands.
// Per-wave tile 32x32 (4 WMMA per 8 ds_read_b128), block tile 128x64.
// ---------------------------------------------------------------------------

typedef __attribute__((ext_vector_type(16))) __bf16 v16bf;
typedef __attribute__((ext_vector_type(8)))  __bf16 v8bf;
typedef __attribute__((ext_vector_type(8)))  float  v8f;

#define D_DIM   2048
#define M_DIM   2048
#define L_SEQ   2048
#define CHUNK_C 128
#define R_ROWS  256
#define NCHUNK  16
#define LR_MEM  0.01f

__device__ __forceinline__ float sigf(float x) { return 1.0f / (1.0f + __expf(-x)); }

enum { EPI_SILU = 0, EPI_DP = 1, EPI_DZ = 2, EPI_WUPD = 3, EPI_OUT = 4 };

// ---------------------------------------------------------------------------
// C(MxN) = epilogue(A' * B'), bf16 operands, block tile 128x64, K step 32.
//   AROWM: A'[m][k]=A[m*lda+k] else A[k*lda+m];  BROWN: B'[k][n]=B[n*ldb+k]
//   else B[k*ldb+n]. 8 waves as 4x2; each wave owns 32x32 (2x2 WMMA tiles).
// ---------------------------------------------------------------------------
template <int EPI, bool AROWM, bool BROWN>
__global__ __launch_bounds__(256) void gemm_wmma(
    const __bf16* __restrict__ A, int lda,
    const __bf16* __restrict__ Bm, int ldb,
    float* __restrict__ C0, __bf16* __restrict__ Cb,
    float* __restrict__ C1, int ldc,
    const float* __restrict__ bias,     // column bias (or nullptr)
    const float* __restrict__ aux,      // DP: Xc ; DZ: Z ; WUPD: {eff_lr,forget}
    int K, float scale, long outRowStrideRows)
{
    __shared__ __bf16 As[128][40];      // [m][k], +8 pad: 16B-aligned rows, no bank repeats
    __shared__ __bf16 Bs[64][40];       // [n][k] so B fragments are contiguous reads

    const int tid  = threadIdx.x;
    const int lane = tid & 31;
    const int wave = tid >> 5;
    const int wr   = wave >> 1;         // 0..3 -> rows [wr*32, +32)
    const int wc   = wave & 1;          // 0..1 -> cols [wc*32, +32)
    const int bm   = blockIdx.y * 128;
    const int bn   = blockIdx.x * 64;
    const int half = lane >> 4;
    const int l15  = lane & 15;

    v8f acc[2][2];
#pragma unroll
    for (int s = 0; s < 2; ++s)
#pragma unroll
        for (int t = 0; t < 2; ++t)
            acc[s][t] = (v8f){0.f, 0.f, 0.f, 0.f, 0.f, 0.f, 0.f, 0.f};

    // staging index precompute
    const int arow = tid >> 1, acolb = (tid & 1) * 16;        // AROWM: 32B per thread
    const int akk  = tid >> 3, amb   = (tid & 7) * 16;        // !AROWM
    const int bn_r = tid >> 2, bcol  = (tid & 3) * 8;         // BROWN: 16B per thread
    const int bkk  = tid >> 3, bnb   = (tid & 7) * 8;         // !BROWN

    const unsigned ldsA = (unsigned)(uintptr_t)&As[arow][acolb];
    const unsigned ldsB = (unsigned)(uintptr_t)&Bs[bn_r][bcol];
    const unsigned long long baseA = (unsigned long long)(uintptr_t)A;
    const unsigned long long baseB = (unsigned long long)(uintptr_t)Bm;

    for (int k0 = 0; k0 < K; k0 += 32) {
        // ---- stage A tile (128 x 32 bf16) ----
        if (AROWM) {
            unsigned goff = ((unsigned)(bm + arow) * (unsigned)lda + (unsigned)(k0 + acolb)) * 2u;
            asm volatile("global_load_async_to_lds_b128 %0, %1, %2"
                         :: "v"(ldsA), "v"(goff), "s"(baseA) : "memory");
            asm volatile("global_load_async_to_lds_b128 %0, %1, %2"
                         :: "v"(ldsA + 16u), "v"(goff + 16u), "s"(baseA) : "memory");
        } else {
            const __bf16* p = &A[(size_t)(k0 + akk) * lda + bm + amb];
            v8bf v0 = *(const v8bf*)p, v1 = *(const v8bf*)(p + 8);
#pragma unroll
            for (int i = 0; i < 8; ++i) { As[amb + i][akk] = v0[i]; As[amb + 8 + i][akk] = v1[i]; }
        }
        // ---- stage B tile (64 x 32 bf16) into Bs[n][k] ----
        if (BROWN) {
            unsigned goff = ((unsigned)(bn + bn_r) * (unsigned)ldb + (unsigned)(k0 + bcol)) * 2u;
            asm volatile("global_load_async_to_lds_b128 %0, %1, %2"
                         :: "v"(ldsB), "v"(goff), "s"(baseB) : "memory");
        } else {
            const __bf16* p = &Bm[(size_t)(k0 + bkk) * ldb + bn + bnb];
            v8bf v0 = *(const v8bf*)p;
#pragma unroll
            for (int i = 0; i < 8; ++i) Bs[bnb + i][bkk] = v0[i];
        }
        if (AROWM || BROWN)
            asm volatile("s_wait_asynccnt 0x0" ::: "memory");
        __syncthreads();

        // ---- fragments: lane holds K = {8h..8h+7} U {16+8h..23+8h} ----
        v16bf a[2], b[2];
#pragma unroll
        for (int s = 0; s < 2; ++s) {
            const int mrow = wr * 32 + s * 16 + l15;
            v8bf lo = *(const v8bf*)&As[mrow][8 * half];
            v8bf hi = *(const v8bf*)&As[mrow][16 + 8 * half];
#pragma unroll
            for (int i = 0; i < 8; ++i) { a[s][i] = lo[i]; a[s][8 + i] = hi[i]; }
        }
#pragma unroll
        for (int t = 0; t < 2; ++t) {
            const int ncol = wc * 32 + t * 16 + l15;
            v8bf lo = *(const v8bf*)&Bs[ncol][8 * half];
            v8bf hi = *(const v8bf*)&Bs[ncol][16 + 8 * half];
#pragma unroll
            for (int i = 0; i < 8; ++i) { b[t][i] = lo[i]; b[t][8 + i] = hi[i]; }
        }

        acc[0][0] = __builtin_amdgcn_wmma_f32_16x16x32_bf16(false, a[0], false, b[0],
                                                            (short)0, acc[0][0], false, false);
        acc[0][1] = __builtin_amdgcn_wmma_f32_16x16x32_bf16(false, a[0], false, b[1],
                                                            (short)0, acc[0][1], false, false);
        acc[1][0] = __builtin_amdgcn_wmma_f32_16x16x32_bf16(false, a[1], false, b[0],
                                                            (short)0, acc[1][0], false, false);
        acc[1][1] = __builtin_amdgcn_wmma_f32_16x16x32_bf16(false, a[1], false, b[1],
                                                            (short)0, acc[1][1], false, false);
        __syncthreads();
    }

    // ---- epilogue: element e of lane L -> row += 8*(L>>4)+e, col = L&15 ----
#pragma unroll
    for (int s = 0; s < 2; ++s) {
#pragma unroll
        for (int t = 0; t < 2; ++t) {
            const int col = bn + wc * 32 + t * 16 + l15;
#pragma unroll
            for (int e = 0; e < 8; ++e) {
                const int row = bm + wr * 32 + s * 16 + half * 8 + e;
                const size_t idx = (size_t)row * ldc + col;
                float v = acc[s][t][e];
                if (EPI == EPI_SILU) {
                    float z = v + bias[col];
                    if (C1) C1[idx] = z;                        // pre-activation
                    Cb[idx] = (__bf16)(z * sigf(z));            // SiLU, bf16 mirror
                } else if (EPI == EPI_DP) {
                    float p = v + bias[col];
                    float d = (p - aux[idx]) * scale;
                    C0[idx] = d; Cb[idx] = (__bf16)d;
                } else if (EPI == EPI_DZ) {
                    float z = aux[idx];
                    float sg = sigf(z);
                    float d = v * (sg * (1.0f + z * (1.0f - sg)));
                    C0[idx] = d; Cb[idx] = (__bf16)d;
                } else if (EPI == EPI_WUPD) {
                    float w = aux[1] * C0[idx] - aux[0] * v;    // forget*W - eff_lr*g
                    C0[idx] = w; Cb[idx] = (__bf16)w;
                } else { // EPI_OUT: scatter to (B,L,D); C0 pre-offset to chunk
                    size_t orow = (size_t)(row >> 7) * outRowStrideRows + (row & 127);
                    C0[orow * (size_t)ldc + col] = v + bias[col];
                }
            }
        }
    }
}

// Gather chunk t (two 128-row blocks L apart) into Xc (f32) + Xb (bf16).
__global__ __launch_bounds__(256) void gather_chunk(const float* __restrict__ src,
                                                    float* __restrict__ dst,
                                                    __bf16* __restrict__ dstb)
{
    const size_t idx = (size_t)blockIdx.x * 256 + threadIdx.x;
    const int r = (int)(idx >> 11), d = (int)(idx & 2047);
    float v = src[(size_t)(r >> 7) * ((size_t)L_SEQ * D_DIM) + (size_t)(r & 127) * D_DIM + d];
    dst[idx] = v;
    dstb[idx] = (__bf16)v;
}

__global__ __launch_bounds__(256) void f32_to_bf16(const float* __restrict__ src,
                                                   __bf16* __restrict__ dst)
{
    const size_t idx = (size_t)blockIdx.x * 256 + threadIdx.x;
    dst[idx] = (__bf16)src[idx];
}

// eff_lr = 0.01*mean_r sig(Xc[r].lr_w + lr_b); forget = mean_b sig(mean_c(Xc[b,c].fg_w)+fg_b)
__global__ __launch_bounds__(256) void chunk_scalars(
    const float* __restrict__ Xc,
    const float* __restrict__ lrw, const float* __restrict__ lrb,
    const float* __restrict__ fgw, const float* __restrict__ fgb,
    float* __restrict__ scal)
{
    const int r = threadIdx.x;
    const float* row = Xc + (size_t)r * D_DIM;
    float dl = 0.f, df = 0.f;
    for (int d = 0; d < D_DIM; ++d) { float xv = row[d]; dl += xv * lrw[d]; df += xv * fgw[d]; }
    __shared__ float slr[256], sfg[256];
    slr[r] = sigf(dl + lrb[0]);
    sfg[r] = df;
    __syncthreads();
    for (int s = 128; s > 0; s >>= 1) { if (r < s) slr[r] += slr[r + s]; __syncthreads(); }
    for (int s = 64;  s > 0; s >>= 1) { if ((r & 127) < s) sfg[r] += sfg[r + s]; __syncthreads(); }
    if (r == 0) {
        scal[0] = LR_MEM * (slr[0] / (float)R_ROWS);
        float t0 = sfg[0] / (float)CHUNK_C, t1 = sfg[128] / (float)CHUNK_C;
        scal[1] = 0.5f * (sigf(t0 + fgb[0]) + sigf(t1 + fgb[0]));
    }
}

__global__ __launch_bounds__(256) void bias_update(
    const float* __restrict__ dP, const float* __restrict__ dZ,
    float* __restrict__ b0, float* __restrict__ b1, const float* __restrict__ scal)
{
    const int j = blockIdx.x * 256 + threadIdx.x;
    const float el = scal[0], fg = scal[1];
    if (j < D_DIM) {
        float g = 0.f;
        for (int r = 0; r < R_ROWS; ++r) g += dP[(size_t)r * D_DIM + j];
        b1[j] = fg * b1[j] - el * g;
    } else {
        const int c = j - D_DIM;
        float g = 0.f;
        for (int r = 0; r < R_ROWS; ++r) g += dZ[(size_t)r * M_DIM + c];
        b0[c] = fg * b0[c] - el * g;
    }
}

extern "C" void kernel_launch(void* const* d_in, const int* in_sizes, int n_in,
                              void* d_out, int out_size, void* d_ws, size_t ws_size,
                              hipStream_t stream)
{
    const float* x   = (const float*)d_in[0];
    const float* W0i = (const float*)d_in[1];
    const float* b0i = (const float*)d_in[2];
    const float* W1i = (const float*)d_in[3];
    const float* b1i = (const float*)d_in[4];
    const float* lrw = (const float*)d_in[5];
    const float* lrb = (const float*)d_in[6];
    const float* fgw = (const float*)d_in[7];
    const float* fgb = (const float*)d_in[8];
    float* out = (float*)d_out;

    // ---- workspace layout: f32 region then bf16 region (~64 MB total) ----
    float* ws = (float*)d_ws;
    size_t o = 0;
    float* W0 = ws + o; o += (size_t)M_DIM * D_DIM;
    float* W1 = ws + o; o += (size_t)D_DIM * M_DIM;
    float* b0 = ws + o; o += M_DIM;
    float* b1 = ws + o; o += D_DIM;
    float* Xc = ws + o; o += (size_t)R_ROWS * D_DIM;
    float* Z  = ws + o; o += (size_t)R_ROWS * M_DIM;
    float* dP = ws + o; o += (size_t)R_ROWS * D_DIM;
    float* dZ = ws + o; o += (size_t)R_ROWS * M_DIM;
    float* scal = ws + o; o += 2;
    o = (o + 7) & ~(size_t)7;                 // 16B-align bf16 region
    __bf16* bws = (__bf16*)(ws + o);
    size_t ob = 0;
    __bf16* W0b = bws + ob; ob += (size_t)M_DIM * D_DIM;
    __bf16* W1b = bws + ob; ob += (size_t)D_DIM * M_DIM;
    __bf16* Xb  = bws + ob; ob += (size_t)R_ROWS * D_DIM;
    __bf16* Hb  = bws + ob; ob += (size_t)R_ROWS * M_DIM;
    __bf16* dPb = bws + ob; ob += (size_t)R_ROWS * D_DIM;
    __bf16* dZb = bws + ob; ob += (size_t)R_ROWS * M_DIM;
    __bf16* Hnb = bws + ob; ob += (size_t)R_ROWS * M_DIM;

    // fresh params every call (deterministic under graph replay)
    hipMemcpyAsync(W0, W0i, (size_t)M_DIM * D_DIM * sizeof(float), hipMemcpyDeviceToDevice, stream);
    hipMemcpyAsync(W1, W1i, (size_t)D_DIM * M_DIM * sizeof(float), hipMemcpyDeviceToDevice, stream);
    hipMemcpyAsync(b0, b0i, (size_t)M_DIM * sizeof(float), hipMemcpyDeviceToDevice, stream);
    hipMemcpyAsync(b1, b1i, (size_t)D_DIM * sizeof(float), hipMemcpyDeviceToDevice, stream);
    const int wElems = M_DIM * D_DIM;
    f32_to_bf16<<<wElems / 256, 256, 0, stream>>>(W0i, W0b);
    f32_to_bf16<<<wElems / 256, 256, 0, stream>>>(W1i, W1b);

    const dim3 blk(256);
    const dim3 gridRN(M_DIM / 64, R_ROWS / 128);   // 256 x 2048 outputs
    const dim3 gridWW(M_DIM / 64, D_DIM / 128);    // 2048 x 2048 outputs
    const float dpScale = 2.0f / (float)((size_t)R_ROWS * D_DIM);

    for (int t = 0; t < NCHUNK; ++t) {
        const float* xch = x + (size_t)t * CHUNK_C * D_DIM;
        gather_chunk<<<(R_ROWS * D_DIM) / 256, blk, 0, stream>>>(xch, Xc, Xb);
        chunk_scalars<<<1, blk, 0, stream>>>(Xc, lrw, lrb, fgw, fgb, scal);

        // Z = Xc@W0^T + b0 ; Hb = silu(Z)
        gemm_wmma<EPI_SILU, true, true><<<gridRN, blk, 0, stream>>>(
            Xb, D_DIM, W0b, D_DIM, nullptr, Hb, Z, M_DIM, b0, nullptr, D_DIM, 0.f, 0);
        // dP = (H@W1^T + b1 - Xc) * scale
        gemm_wmma<EPI_DP, true, true><<<gridRN, blk, 0, stream>>>(
            Hb, M_DIM, W1b, M_DIM, dP, dPb, nullptr, D_DIM, b1, Xc, M_DIM, dpScale, 0);
        // dZ = (dP@W1) * silu'(Z)
        gemm_wmma<EPI_DZ, true, false><<<gridRN, blk, 0, stream>>>(
            dPb, D_DIM, W1b, M_DIM, dZ, dZb, nullptr, M_DIM, nullptr, Z, D_DIM, 0.f, 0);
        // bias updates
        bias_update<<<(D_DIM + M_DIM) / 256, blk, 0, stream>>>(dP, dZ, b0, b1, scal);
        // W1 <- fg*W1 - lr * dP^T @ H   (old W1 fully consumed above)
        gemm_wmma<EPI_WUPD, false, false><<<gridWW, blk, 0, stream>>>(
            dPb, D_DIM, Hb, M_DIM, W1, W1b, nullptr, M_DIM, nullptr, scal, R_ROWS, 0.f, 0);
        // W0 <- fg*W0 - lr * dZ^T @ Xc
        gemm_wmma<EPI_WUPD, false, false><<<gridWW, blk, 0, stream>>>(
            dZb, M_DIM, Xb, D_DIM, W0, W0b, nullptr, D_DIM, nullptr, scal, R_ROWS, 0.f, 0);
        // re-forward through updated params
        gemm_wmma<EPI_SILU, true, true><<<gridRN, blk, 0, stream>>>(
            Xb, D_DIM, W0b, D_DIM, nullptr, Hnb, nullptr, M_DIM, b0, nullptr, D_DIM, 0.f, 0);
        // out_chunk = Hn@W1^T + b1, scattered to (B, L, D)
        gemm_wmma<EPI_OUT, true, true><<<gridRN, blk, 0, stream>>>(
            Hnb, M_DIM, W1b, M_DIM, out + (size_t)t * CHUNK_C * D_DIM, nullptr, nullptr, D_DIM,
            b1, nullptr, M_DIM, 0.f, (long)L_SEQ);
    }
    (void)in_sizes; (void)n_in; (void)out_size; (void)ws_size;
}
